// QuantizedLinear_1692217114910
// MI455X (gfx1250) — matrix operand order repaired and checked by reading, not compiled, not run on hardware
//
#include <hip/hip_runtime.h>

typedef __attribute__((ext_vector_type(16))) _Float16 v16h;
typedef __attribute__((ext_vector_type(8)))  _Float16 v8h;
typedef __attribute__((ext_vector_type(8)))  float    v8f;

#define M_DIM 8192   // B*S = 4*2048
#define N_DIM 4096   // D_OUT
#define K_DIM 4096   // D_IN
#define BM 128
#define BN 128
#define BK 32
#define LDSK 40      // BK + 8 halfs pad: 80-byte row stride, keeps 16B alignment

// One K-tile worth of per-thread staging registers (16 x-floats + 16 q-ints)
struct Tile {
    float4 x[4];
    int4   q[4];
};

static __device__ __forceinline__ void load_tile(Tile& t, const float* __restrict__ xg,
                                                 const int* __restrict__ wg, int k0) {
#pragma unroll
    for (int i = 0; i < 4; ++i) {
        t.x[i] = *(const float4*)(xg + k0 + 4 * i);
        t.q[i] = *(const int4*)(wg + k0 + 4 * i);
    }
}

// pack two float4 -> 8 x f16
static __device__ __forceinline__ v8h cvt8(const float4 a, const float4 b) {
    v8h r;
    r[0] = (_Float16)a.x; r[1] = (_Float16)a.y; r[2] = (_Float16)a.z; r[3] = (_Float16)a.w;
    r[4] = (_Float16)b.x; r[5] = (_Float16)b.y; r[6] = (_Float16)b.z; r[7] = (_Float16)b.w;
    return r;
}

// dequantize two int4 -> 8 x f16:  q*scale + (-zero*scale)  (one v_fma each)
static __device__ __forceinline__ v8h dq8(const int4 a, const int4 b,
                                          const float sc, const float nzs) {
    v8h r;
    r[0] = (_Float16)__builtin_fmaf((float)a.x, sc, nzs);
    r[1] = (_Float16)__builtin_fmaf((float)a.y, sc, nzs);
    r[2] = (_Float16)__builtin_fmaf((float)a.z, sc, nzs);
    r[3] = (_Float16)__builtin_fmaf((float)a.w, sc, nzs);
    r[4] = (_Float16)__builtin_fmaf((float)b.x, sc, nzs);
    r[5] = (_Float16)__builtin_fmaf((float)b.y, sc, nzs);
    r[6] = (_Float16)__builtin_fmaf((float)b.z, sc, nzs);
    r[7] = (_Float16)__builtin_fmaf((float)b.w, sc, nzs);
    return r;
}

// convert/dequantize one staged tile into LDS
static __device__ __forceinline__ void stage(const Tile& t, v8h* ad, v8h* bd,
                                             const float sc, const float nzs) {
    ad[0] = cvt8(t.x[0], t.x[1]);
    ad[1] = cvt8(t.x[2], t.x[3]);
    bd[0] = dq8(t.q[0], t.q[1], sc, nzs);
    bd[1] = dq8(t.q[2], t.q[3], sc, nzs);
}

// fragment loads + 8 WMMAs for one BK=32 slab
static __device__ __forceinline__ void mma_step(v8f (&acc)[4][2],
                                                const _Float16* __restrict__ As,
                                                const _Float16* __restrict__ Bs,
                                                int wm, int wn, int g, int ln) {
    // A fragments: lane holds row m=ln; contiguous runs at k = g*8 and 16+g*8
    v16h afrag[4];
#pragma unroll
    for (int i = 0; i < 4; ++i) {
        const _Float16* ap = &As[(wm * 64 + i * 16 + ln) * LDSK];
        v8h lo = *(const v8h*)(ap + g * 8);
        v8h hi = *(const v8h*)(ap + 16 + g * 8);
        afrag[i] = __builtin_shufflevector(lo, hi,
                    0,1,2,3,4,5,6,7,8,9,10,11,12,13,14,15);
    }
    // B fragments: lane holds col n=ln; contiguous runs at k = g*8 and 16+g*8
    v16h bfrag[2];
#pragma unroll
    for (int j = 0; j < 2; ++j) {
        const _Float16* bp = &Bs[(wn * 32 + j * 16 + ln) * LDSK + g * 8];
        v8h lo = *(const v8h*)(bp);
        v8h hi = *(const v8h*)(bp + 16);
        bfrag[j] = __builtin_shufflevector(lo, hi,
                    0,1,2,3,4,5,6,7,8,9,10,11,12,13,14,15);
    }
#pragma unroll
    for (int i = 0; i < 4; ++i)
#pragma unroll
        for (int j = 0; j < 2; ++j)
            acc[i][j] = __builtin_amdgcn_wmma_f32_16x16x32_f16(
                false, afrag[i], false, bfrag[j],
                (short)0, acc[i][j], false, false);
}

__global__ __launch_bounds__(256, 1) void qlinear_wmma_f16(
    const float* __restrict__ X,       // [M_DIM, K_DIM] fp32
    const int*   __restrict__ Wq,      // [N_DIM, K_DIM] int32 (int8 range)
    const float* __restrict__ scales,  // [N_DIM]
    const float* __restrict__ zeros,   // [N_DIM]
    const float* __restrict__ bias,    // [N_DIM]
    float*       __restrict__ Y)       // [M_DIM, N_DIM] fp32
{
    __shared__ __align__(16) _Float16 As[BM * LDSK];
    __shared__ __align__(16) _Float16 Bs[BN * LDSK];

    const int tid  = threadIdx.x;
    const int lane = tid & 31;
    const int wid  = tid >> 5;      // 0..7
    const int wm   = wid >> 2;      // 0..1  : wave row   (64 rows of C each)
    const int wn   = wid & 3;       // 0..3  : wave col   (32 cols of C each)
    const int g    = lane >> 4;     // K-half selector per WMMA VGPR layout
    const int ln   = lane & 15;

    const int bm = blockIdx.y * BM;
    const int bn = blockIdx.x * BN;

    // Staging: 2 threads per tile row, 16 elements each
    const int srow = tid >> 1;           // 0..127
    const int scol = (tid & 1) << 4;     // 0 or 16

    const float sc  = scales[bn + srow];
    const float nzs = -zeros[bn + srow] * sc;

    const float* xg = X  + (size_t)(bm + srow) * K_DIM + scol;
    const int*   wg = Wq + (size_t)(bn + srow) * K_DIM + scol;

    v8h* const ad = (v8h*)&As[srow * LDSK + scol];
    v8h* const bd = (v8h*)&Bs[srow * LDSK + scol];

    v8f acc[4][2] = {};   // 8 x 16x16 f32 tiles per wave (64 VGPRs)

    // ---- software pipeline: two explicit register tiles (no spillable array) ----
    Tile t0, t1;
    load_tile(t0, xg, wg, 0);

    for (int k0 = 0; k0 < K_DIM; k0 += 2 * BK) {
        // ---------------- first BK slab (t0) ----------------
        stage(t0, ad, bd, sc, nzs);
        __syncthreads();

        // next tile's globals fly while the WMMAs below execute
        load_tile(t1, xg, wg, k0 + BK);          // k0+BK <= K-32: always in-bounds
        if (k0 + 2 * BK < K_DIM) {
            __builtin_prefetch(xg + k0 + 2 * BK, 0, 1);   // global_prefetch_b8
            __builtin_prefetch(wg + k0 + 2 * BK, 0, 1);
        }

        mma_step(acc, As, Bs, wm, wn, g, ln);
        __syncthreads();

        // ---------------- second BK slab (t1) ----------------
        stage(t1, ad, bd, sc, nzs);
        __syncthreads();

        if (k0 + 2 * BK < K_DIM) {
            load_tile(t0, xg, wg, k0 + 2 * BK);
            if (k0 + 3 * BK < K_DIM) {
                __builtin_prefetch(xg + k0 + 3 * BK, 0, 1);
                __builtin_prefetch(wg + k0 + 3 * BK, 0, 1);
            }
        }

        mma_step(acc, As, Bs, wm, wn, g, ln);
        __syncthreads();
    }

    // ---- epilogue: bias add + store (C layout: M = 8*g + r, N = ln) ----
#pragma unroll
    for (int j = 0; j < 2; ++j) {
        const int n = bn + wn * 32 + j * 16 + ln;
        const float bv = bias[n];
#pragma unroll
        for (int i = 0; i < 4; ++i) {
            const int m0 = bm + wm * 64 + i * 16 + g * 8;
            float* yp = Y + (size_t)m0 * N_DIM + n;
#pragma unroll
            for (int r = 0; r < 8; ++r)
                yp[(size_t)r * N_DIM] = acc[i][j][r] + bv;
        }
    }
}

extern "C" void kernel_launch(void* const* d_in, const int* in_sizes, int n_in,
                              void* d_out, int out_size, void* d_ws, size_t ws_size,
                              hipStream_t stream) {
    const float* x      = (const float*)d_in[0];  // [4,2048,4096] fp32
    const int*   wq     = (const int*)  d_in[1];  // [4096,4096] int32
    const float* scales = (const float*)d_in[2];  // [4096,1]
    const float* zeros  = (const float*)d_in[3];  // [4096,1]
    const float* bias   = (const float*)d_in[4];  // [4096]
    float*       y      = (float*)d_out;          // [4,2048,4096] fp32

    dim3 grid(N_DIM / BN, M_DIM / BM);   // (32, 64)
    qlinear_wmma_f16<<<grid, dim3(256), 0, stream>>>(x, wq, scales, zeros, bias, y);
}
